// PatchMerging_7095285973442
// MI455X (gfx1250) — compile-verified, matrix operand order checked by reading, not run on hardware
//
#include <hip/hip_runtime.h>
#include <hip/hip_bf16.h>

// ---------------------------------------------------------------------------
// MI455X (gfx1250) implementation. wave32, WMMA bf16 16x16x32 for all GEMMs,
// async global->LDS tile staging (ASYNCcnt double-buffer), fused flash
// attention, scalar VALU for the bandwidth-trivial glue.
// ---------------------------------------------------------------------------

typedef __attribute__((ext_vector_type(16))) __bf16 v16bf;
typedef __attribute__((ext_vector_type(8)))  __bf16 v8bf;
typedef __attribute__((ext_vector_type(8)))  float  v8f;

typedef unsigned short u16;
typedef unsigned int   u32;

#define NB   8
#define LL   4096
#define DIMC 256
#define EPSV 1e-5f
#define SCALE_ATTN 0.125f   // (256/4)^-0.5

// ---- helpers ---------------------------------------------------------------

__device__ __forceinline__ u16 f2bf_bits(float f) {
  union { float f; u32 u; } x; x.f = f;
  u32 u = x.u;
  u32 r = (u + 0x7FFFu + ((u >> 16) & 1u)) >> 16;   // round-to-nearest-even
  return (u16)r;
}
__device__ __forceinline__ float bf_bits2f(u16 b) {
  union { u32 u; float f; } x; x.u = ((u32)b) << 16;
  return x.f;
}
__device__ __forceinline__ float gelu_erf(float x) {
  return 0.5f * x * (1.0f + erff(x * 0.70710678118654752f));
}

__device__ __forceinline__ v8f wmma_bf16(v16bf a, v16bf b, v8f c) {
  return __builtin_amdgcn_wmma_f32_16x16x32_bf16(false, a, false, b,
                                                 (short)0, c, false, false);
}

// CDNA5 async copy: global -> LDS, 16 bytes per lane, tracked by ASYNCcnt.
// lds_off is the LDS byte offset (low 32 bits of the flat shared address).
__device__ __forceinline__ void async_cp_b128(u32 lds_off, const void* gptr) {
  asm volatile("global_load_async_to_lds_b128 %0, %1, off"
               :: "v"(lds_off), "v"((unsigned long long)(size_t)gptr)
               : "memory");
}

// Load a 16x32 bf16 fragment (A layout: lanes 0-15 = rows, K chunks
// {h*8..h*8+7} and {16+h*8..}).  Same pattern serves B fragments when the
// tile is stored [n][k] with k contiguous.
__device__ __forceinline__ v16bf frag_ld(const u16* base, int stride, int koff) {
  const int lane = threadIdx.x & 31;
  const int m = lane & 15, h = lane >> 4;
  const u16* r = base + (size_t)m * stride + koff;
  v8bf a = *(const v8bf*)(r + h * 8);
  v8bf b = *(const v8bf*)(r + 16 + h * 8);
  return __builtin_shufflevector(a, b, 0,1,2,3,4,5,6,7,8,9,10,11,12,13,14,15);
}

// ---- generic WMMA GEMM: out = act(A[M,K] * W[N,K]^T + bias) ---------------
// block = 256 threads (8 waves, 4M x 2N), block tile 128x128, wave tile
// 32x64 (8 WMMA per K-step, 6 fragment loads).  Double-buffered async
// global->LDS staging: each lane issues exactly 4 async b128 per tile.

#define AST 48   // LDS row stride in halfs (16B-aligned chunks, conflict pad)

__global__ __launch_bounds__(256) void k_gemm_bf16(
    const u16* __restrict__ A, const u16* __restrict__ Wt,
    const float* __restrict__ bias,
    float* __restrict__ outF, u16* __restrict__ outB,
    int M, int N, int K, int act)
{
  __shared__ u16 As[2][128 * AST];
  __shared__ u16 Bs[2][128 * AST];

  const int tid  = threadIdx.x;
  const int lane = tid & 31, wv = tid >> 5;
  const int wm = wv & 3, wn = wv >> 2;       // 4 x 2 wave grid
  const int m0 = blockIdx.y * 128;
  const int n0 = blockIdx.x * 128;

  const u32 aoff0 = (u32)(size_t)&As[0][0];
  const u32 aoff1 = (u32)(size_t)&As[1][0];
  const u32 boff0 = (u32)(size_t)&Bs[0][0];
  const u32 boff1 = (u32)(size_t)&Bs[1][0];

  v8f acc[2][4];
#pragma unroll
  for (int i = 0; i < 2; ++i)
#pragma unroll
    for (int j = 0; j < 4; ++j)
      acc[i][j] = (v8f){0.f,0.f,0.f,0.f,0.f,0.f,0.f,0.f};

  const int nk = K >> 5;

  auto stage = [&](int kt, int bufi) {
    const int k0 = kt << 5;
    const u32 ao = bufi ? aoff1 : aoff0;
    const u32 bo = bufi ? boff1 : boff0;
#pragma unroll
    for (int j = 0; j < 2; ++j) {
      const int id  = tid * 2 + j;     // 0..511
      const int row = id >> 2;         // 0..127
      const int col = (id & 3) * 8;    // halfs
      async_cp_b128(ao + (u32)(row * AST + col) * 2,
                    A  + (size_t)(m0 + row) * K + k0 + col);
      async_cp_b128(bo + (u32)(row * AST + col) * 2,
                    Wt + (size_t)(n0 + row) * K + k0 + col);
    }
  };

  stage(0, 0);

  for (int kt = 0; kt < nk; ++kt) {
    const int cur = kt & 1;
    if (kt + 1 < nk) {
      stage(kt + 1, cur ^ 1);
      asm volatile("s_wait_asynccnt 0x4" ::: "memory");  // tile kt landed
    } else {
      asm volatile("s_wait_asynccnt 0x0" ::: "memory");
    }
    __syncthreads();

    const u16* Ab = &As[cur][0];
    const u16* Bb = &Bs[cur][0];
    v16bf a0 = frag_ld(Ab + (wm * 32     ) * AST, AST, 0);
    v16bf a1 = frag_ld(Ab + (wm * 32 + 16) * AST, AST, 0);
#pragma unroll
    for (int sub = 0; sub < 4; ++sub) {
      v16bf bf = frag_ld(Bb + (wn * 64 + sub * 16) * AST, AST, 0);
      acc[0][sub] = wmma_bf16(a0, bf, acc[0][sub]);
      acc[1][sub] = wmma_bf16(a1, bf, acc[1][sub]);
    }
    __syncthreads();   // all waves done reading before buffer reuse
  }

  const int h = lane >> 4, cn = lane & 15;
#pragma unroll
  for (int ms = 0; ms < 2; ++ms) {
#pragma unroll
    for (int sub = 0; sub < 4; ++sub) {
      v8f a = acc[ms][sub];
      const int col = n0 + wn * 64 + sub * 16 + cn;
      const float bval = bias ? bias[col] : 0.f;
#pragma unroll
      for (int r = 0; r < 8; ++r) {
        const int row = m0 + wm * 32 + ms * 16 + h * 8 + r;
        float v = a[r] + bval;
        if (act == 1) v = gelu_erf(v);
        const size_t o = (size_t)row * N + col;
        if (outF) outF[o] = v;
        if (outB) outB[o] = f2bf_bits(v);
      }
    }
  }
}

// ---- fused cross-shaped-window flash attention -----------------------------
// block: 256 threads = 8 waves; each wave owns 16 q-rows; block = 128 q rows.
// blockIdx.x bits: [0]=branch [3:1]=batch [6:4]=win [7]=head [9:8]=q-chunk

#define CST 72   // LDS row stride (halfs), 16B-aligned chunks, reduced conflicts

__global__ __launch_bounds__(256) void k_attn(
    const u16* __restrict__ qkv, float* __restrict__ out)
{
  __shared__ u16 Ks[64 * CST];        // [token][feat]
  __shared__ u16 Vs[64 * CST];        // [feat][token]  (transposed)
  __shared__ u16 Ps[8][16 * CST];     // per-wave P tile [row][token]

  const int tid  = threadIdx.x;
  const int lane = tid & 31, wv = tid >> 5;
  const int h = lane >> 4, ln16 = lane & 15;

  const int bi     = blockIdx.x;
  const int branch = bi & 1;
  const int b      = (bi >> 1) & 7;
  const int win    = (bi >> 4) & 7;
  const int head   = (bi >> 7) & 1;
  const int qc     = (bi >> 8) & 3;

  const int cb = branch * 128 + head * 64;
  const int p0 = qc * 128 + wv * 16;

  auto tok2l = [&](int p) -> int {
    if (branch == 0) return ((p >> 3) << 6) + (win << 3) + (p & 7);
    else             return (((win << 3) + (p >> 6)) << 6) + (p & 63);
  };

  // Q fragments: 2 x (16x32) covering the 64 head features
  v16bf qf[2];
  {
    const int p = p0 + ln16;
    const int l = tok2l(p);
    const u16* qb = qkv + ((size_t)(b * LL + l) * 768) + cb;
#pragma unroll
    for (int kk = 0; kk < 2; ++kk) {
      v8bf a = *(const v8bf*)(qb + kk * 32 + h * 8);
      v8bf c = *(const v8bf*)(qb + kk * 32 + 16 + h * 8);
      qf[kk] = __builtin_shufflevector(a, c, 0,1,2,3,4,5,6,7,8,9,10,11,12,13,14,15);
    }
  }

  v8f z = {0.f,0.f,0.f,0.f,0.f,0.f,0.f,0.f};
  v8f ofr[4];
#pragma unroll
  for (int i = 0; i < 4; ++i) ofr[i] = z;
  float mrow[8], lrow[8];
#pragma unroll
  for (int r = 0; r < 8; ++r) { mrow[r] = -1e30f; lrow[r] = 0.f; }

  for (int kt0 = 0; kt0 < 512; kt0 += 64) {
    // stage K chunk [token][feat]
    {
      const int tk  = tid >> 2;
      const int seg = (tid & 3) * 16;
      const int l = tok2l(kt0 + tk);
      const u16* kb = qkv + ((size_t)(b * LL + l) * 768) + 256 + cb;
      *(float4*)(Ks + tk * CST + seg)     = *(const float4*)(kb + seg);
      *(float4*)(Ks + tk * CST + seg + 8) = *(const float4*)(kb + seg + 8);
    }
    // stage V chunk transposed [feat][token]
#pragma unroll 4
    for (int i = 0; i < 16; ++i) {
      const int idx = tid * 16 + i;
      const int tk2 = idx >> 6, ch = idx & 63;
      const int l2 = tok2l(kt0 + tk2);
      Vs[ch * CST + tk2] =
          qkv[((size_t)(b * LL + l2) * 768) + 512 + cb + ch];
    }
    __syncthreads();

    // S = Q K^T  (4 col-tiles x 2 k-steps = 8 WMMA)
    v8f sfr[4];
#pragma unroll
    for (int ns = 0; ns < 4; ++ns) sfr[ns] = z;
#pragma unroll
    for (int kk = 0; kk < 2; ++kk)
#pragma unroll
      for (int ns = 0; ns < 4; ++ns)
        sfr[ns] = wmma_bf16(qf[kk], frag_ld(Ks + (ns * 16) * CST, CST, kk * 32),
                            sfr[ns]);

    // online softmax (row stats live per half-wave, width-16 shuffles)
#pragma unroll
    for (int r = 0; r < 8; ++r) {
      float sv0 = sfr[0][r] * SCALE_ATTN;
      float sv1 = sfr[1][r] * SCALE_ATTN;
      float sv2 = sfr[2][r] * SCALE_ATTN;
      float sv3 = sfr[3][r] * SCALE_ATTN;
      float mx = fmaxf(fmaxf(sv0, sv1), fmaxf(sv2, sv3));
#pragma unroll
      for (int o = 1; o < 16; o <<= 1) mx = fmaxf(mx, __shfl_xor(mx, o, 16));
      const float mn = fmaxf(mrow[r], mx);
      const float corr = __expf(mrow[r] - mn);
      const float p0e = __expf(sv0 - mn);
      const float p1e = __expf(sv1 - mn);
      const float p2e = __expf(sv2 - mn);
      const float p3e = __expf(sv3 - mn);
      float rs = p0e + p1e + p2e + p3e;
#pragma unroll
      for (int o = 1; o < 16; o <<= 1) rs += __shfl_xor(rs, o, 16);
      mrow[r] = mn;
      lrow[r] = lrow[r] * corr + rs;
      ofr[0][r] *= corr; ofr[1][r] *= corr;
      ofr[2][r] *= corr; ofr[3][r] *= corr;
      const int row = h * 8 + r;
      Ps[wv][row * CST +  0 + ln16] = f2bf_bits(p0e);
      Ps[wv][row * CST + 16 + ln16] = f2bf_bits(p1e);
      Ps[wv][row * CST + 32 + ln16] = f2bf_bits(p2e);
      Ps[wv][row * CST + 48 + ln16] = f2bf_bits(p3e);
    }

    // O += P V   (per-wave P in LDS; V transposed so B-frag k is contiguous)
#pragma unroll
    for (int kk = 0; kk < 2; ++kk) {
      v16bf a = frag_ld(&Ps[wv][0], CST, kk * 32);
#pragma unroll
      for (int ns = 0; ns < 4; ++ns)
        ofr[ns] = wmma_bf16(a, frag_ld(Vs + (ns * 16) * CST, CST, kk * 32),
                            ofr[ns]);
    }
    __syncthreads();
  }

  // normalize + scatter back in windows2img order
#pragma unroll
  for (int ns = 0; ns < 4; ++ns) {
    const int ch = cb + ns * 16 + ln16;
#pragma unroll
    for (int r = 0; r < 8; ++r) {
      const int p = p0 + h * 8 + r;
      const int l = tok2l(p);
      out[((size_t)(b * LL + l)) * DIMC + ch] = ofr[ns][r] / lrow[r];
    }
  }
}

// ---- scalar glue kernels ---------------------------------------------------

__global__ void k_f2bf(const float* __restrict__ in, u16* __restrict__ out, int n) {
  int i = blockIdx.x * 256 + threadIdx.x;
  if (i < n) out[i] = f2bf_bits(in[i]);
}

// patch embed: conv4x4 s4 + GELU + BN -> x3 bf16 [B*L, 256]
__global__ void k_embed(const float* __restrict__ x,
                        const float* __restrict__ cw, const float* __restrict__ cbias,
                        const float* __restrict__ g, const float* __restrict__ be,
                        const float* __restrict__ mn, const float* __restrict__ vr,
                        u16* __restrict__ x3)
{
  size_t idx = (size_t)blockIdx.x * 256 + threadIdx.x;   // B*L*256
  const int c = idx & 255;
  const int l = (idx >> 8) & 4095;
  const int b = idx >> 20;
  const int hh = l >> 6, ww = l & 63;
  const float* xb = x + ((size_t)b * 256 + hh * 4) * 256 + ww * 4;
  float s = cbias[c];
#pragma unroll
  for (int p = 0; p < 4; ++p)
#pragma unroll
    for (int q = 0; q < 4; ++q)
      s += xb[p * 256 + q] * cw[c * 16 + p * 4 + q];
  s = gelu_erf(s);
  s = (s - mn[c]) * rsqrtf(vr[c] + EPSV) * g[c] + be[c];
  x3[idx] = f2bf_bits(s);
}

// LePE depthwise 3x3 within each cross-shaped window, added into attn output
__global__ void k_lepe(const u16* __restrict__ qkv,
                       const float* __restrict__ w0, const float* __restrict__ b0,
                       const float* __restrict__ w1, const float* __restrict__ b1,
                       float* __restrict__ out)
{
  size_t idx = (size_t)blockIdx.x * 256 + threadIdx.x;   // B*L*256
  const int ch = idx & 255;
  const int l  = (idx >> 8) & 4095;
  const int b  = idx >> 20;
  const int branch = ch >> 7, cc = ch & 127;
  const int hh = l >> 6, ww = l & 63;
  const float* wt = (branch ? w1 : w0) + cc * 9;
  float acc = (branch ? b1 : b0)[cc];
  int sh, sw, Hsp, Wsp;
  if (branch == 0) { Hsp = 64; Wsp = 8;  sh = hh;     sw = ww & 7; }
  else             { Hsp = 8;  Wsp = 64; sh = hh & 7; sw = ww;     }
#pragma unroll
  for (int dh = -1; dh <= 1; ++dh)
#pragma unroll
    for (int dw = -1; dw <= 1; ++dw) {
      const int nh = sh + dh, nw = sw + dw;
      if (nh < 0 || nh >= Hsp || nw < 0 || nw >= Wsp) continue;
      int nl;
      if (branch == 0) nl = (nh << 6) + (ww & ~7) + nw;
      else             nl = (((hh & ~7) + nh) << 6) + nw;
      const float v = bf_bits2f(
          qkv[((size_t)(b * LL + nl)) * 768 + 512 + branch * 128 + cc]);
      acc += v * wt[(dh + 1) * 3 + (dw + 1)];
    }
  out[idx] += acc;
}

// LayerNorm over rows of 256, bf16 out
__global__ __launch_bounds__(256) void k_ln256(
    const float* __restrict__ in, const float* __restrict__ g,
    const float* __restrict__ b, u16* __restrict__ out)
{
  __shared__ float ss[8], qq[8];
  const int row = blockIdx.x;
  const float v = in[(size_t)row * 256 + threadIdx.x];
  float s = v, q = v * v;
#pragma unroll
  for (int o = 16; o; o >>= 1) { s += __shfl_xor(s, o, 32); q += __shfl_xor(q, o, 32); }
  if ((threadIdx.x & 31) == 0) { ss[threadIdx.x >> 5] = s; qq[threadIdx.x >> 5] = q; }
  __syncthreads();
  if (threadIdx.x == 0) {
    float S = 0, Q = 0;
    for (int i = 0; i < 8; ++i) { S += ss[i]; Q += qq[i]; }
    const float m = S / 256.f;
    ss[0] = m; qq[0] = Q / 256.f - m * m;
  }
  __syncthreads();
  const float m = ss[0], var = qq[0];
  const float r = (v - m) * rsqrtf(var + EPSV) * g[threadIdx.x] + b[threadIdx.x];
  out[(size_t)row * 256 + threadIdx.x] = f2bf_bits(r);
}

// window-reverse + 2x2 patch-merge gather + LayerNorm(1024), bf16 out
__global__ __launch_bounds__(256) void k_merge_ln(
    const float* __restrict__ x6f, const float* __restrict__ g,
    const float* __restrict__ be, u16* __restrict__ out)
{
  __shared__ float ss[8], qq[8];
  const int row = blockIdx.x;               // b*1024 + h2*32 + w2
  const int w2 = row & 31, h2 = (row >> 5) & 31, b = row >> 10;
  float vals[4];
#pragma unroll
  for (int i = 0; i < 4; ++i) {
    const int hh = 2 * h2 + ((i == 1) || (i == 3));
    const int ww = 2 * w2 + (i >= 2);
    const int l6 = (((hh >> 2) * 16 + (ww >> 2)) << 4) + (hh & 3) * 4 + (ww & 3);
    vals[i] = x6f[((size_t)(b * LL + l6)) * 256 + threadIdx.x];
  }
  float s = vals[0] + vals[1] + vals[2] + vals[3];
  float q = vals[0]*vals[0] + vals[1]*vals[1] + vals[2]*vals[2] + vals[3]*vals[3];
#pragma unroll
  for (int o = 16; o; o >>= 1) { s += __shfl_xor(s, o, 32); q += __shfl_xor(q, o, 32); }
  if ((threadIdx.x & 31) == 0) { ss[threadIdx.x >> 5] = s; qq[threadIdx.x >> 5] = q; }
  __syncthreads();
  if (threadIdx.x == 0) {
    float S = 0, Q = 0;
    for (int i = 0; i < 8; ++i) { S += ss[i]; Q += qq[i]; }
    const float m = S / 1024.f;
    ss[0] = m; qq[0] = Q / 1024.f - m * m;
  }
  __syncthreads();
  const float m = ss[0], rs = rsqrtf(qq[0] + EPSV);
#pragma unroll
  for (int i = 0; i < 4; ++i) {
    const int c = i * 256 + threadIdx.x;
    out[(size_t)row * 1024 + c] = f2bf_bits((vals[i] - m) * rs * g[c] + be[c]);
  }
}

// deconv 4x4 s4 (k==s scatter) + GELU + BN -> final output [8,128,128]
__global__ void k_deconv(const float* __restrict__ x11, const float* __restrict__ dw,
                         const float* __restrict__ db, const float* __restrict__ g,
                         const float* __restrict__ be, const float* __restrict__ mn,
                         const float* __restrict__ vr, float* __restrict__ out)
{
  const size_t idx = (size_t)blockIdx.x * 256 + threadIdx.x;  // 8*128*128
  const int ww = idx & 127, hh = (idx >> 7) & 127, b = idx >> 14;
  const int h2 = hh >> 2, p = hh & 3, w2 = ww >> 2, q = ww & 3;
  const float* xr = x11 + ((size_t)((b * 32 + h2) * 32 + w2)) * 512;
  const int wo = p * 4 + q;
  float s = db[0];
  for (int c = 0; c < 512; ++c) s += xr[c] * dw[c * 16 + wo];
  s = gelu_erf(s);
  s = (s - mn[0]) * rsqrtf(vr[0] + EPSV) * g[0] + be[0];
  out[idx] = s;
}

// ---------------------------------------------------------------------------

extern "C" void kernel_launch(void* const* d_in, const int* in_sizes, int n_in,
                              void* d_out, int out_size, void* d_ws, size_t ws_size,
                              hipStream_t stream) {
  (void)in_sizes; (void)n_in; (void)out_size; (void)ws_size;
  const float* x       = (const float*)d_in[0];
  const float* conv1_w = (const float*)d_in[2];
  const float* conv1_b = (const float*)d_in[3];
  const float* bn1_g   = (const float*)d_in[4];
  const float* bn1_b   = (const float*)d_in[5];
  const float* bn1_m   = (const float*)d_in[6];
  const float* bn1_v   = (const float*)d_in[7];
  const float* qkv_w   = (const float*)d_in[8];
  const float* lepe_w0 = (const float*)d_in[9];
  const float* lepe_b0 = (const float*)d_in[10];
  const float* lepe_w1 = (const float*)d_in[11];
  const float* lepe_b1 = (const float*)d_in[12];
  const float* proj_w  = (const float*)d_in[13];
  const float* proj_b  = (const float*)d_in[14];
  const float* ln1_g   = (const float*)d_in[15];
  const float* ln1_b   = (const float*)d_in[16];
  const float* fc1_w   = (const float*)d_in[17];
  const float* fc1_b   = (const float*)d_in[18];
  const float* fc2_w   = (const float*)d_in[19];
  const float* fc2_b   = (const float*)d_in[20];
  const float* ln_g    = (const float*)d_in[21];
  const float* ln_b    = (const float*)d_in[22];
  const float* red_w   = (const float*)d_in[23];
  const float* red_b   = (const float*)d_in[24];
  const float* dec_w   = (const float*)d_in[25];
  const float* dec_b   = (const float*)d_in[26];
  const float* bn2_g   = (const float*)d_in[27];
  const float* bn2_b   = (const float*)d_in[28];
  const float* bn2_m   = (const float*)d_in[29];
  const float* bn2_v   = (const float*)d_in[30];

  char* wp = (char*)d_ws;
  auto carve = [&](size_t bytes) -> void* {
    void* p = (void*)wp; wp += (bytes + 255) & ~(size_t)255; return p;
  };
  u16*   qkvw_b = (u16*)carve((size_t)768 * 256 * 2);
  u16*   projw_b= (u16*)carve((size_t)256 * 256 * 2);
  u16*   fc1w_b = (u16*)carve((size_t)1024 * 256 * 2);
  u16*   fc2w_b = (u16*)carve((size_t)256 * 1024 * 2);
  u16*   redw_b = (u16*)carve((size_t)512 * 1024 * 2);
  u16*   x3b    = (u16*)carve((size_t)32768 * 256 * 2);
  u16*   qkvb   = (u16*)carve((size_t)32768 * 768 * 2);
  float* x6f    = (float*)carve((size_t)32768 * 256 * 4);
  u16*   x6b    = (u16*)carve((size_t)32768 * 256 * 2);
  float* projf  = (float*)carve((size_t)32768 * 256 * 4);
  u16*   lnb    = (u16*)carve((size_t)32768 * 256 * 2);
  u16*   fc1o   = (u16*)carve((size_t)32768 * 1024 * 2);
  float* x6ff   = (float*)carve((size_t)32768 * 256 * 4);
  u16*   x9b    = (u16*)carve((size_t)8192 * 1024 * 2);
  float* x11    = (float*)carve((size_t)8192 * 512 * 4);

  // weights -> bf16
  k_f2bf<<<(768*256+255)/256, 256, 0, stream>>>(qkv_w, qkvw_b, 768*256);
  k_f2bf<<<(256*256+255)/256, 256, 0, stream>>>(proj_w, projw_b, 256*256);
  k_f2bf<<<(1024*256+255)/256,256, 0, stream>>>(fc1_w, fc1w_b, 1024*256);
  k_f2bf<<<(256*1024+255)/256,256, 0, stream>>>(fc2_w, fc2w_b, 256*1024);
  k_f2bf<<<(512*1024+255)/256,256, 0, stream>>>(red_w, redw_b, 512*1024);

  // patch embed
  k_embed<<<32768, 256, 0, stream>>>(x, conv1_w, conv1_b, bn1_g, bn1_b,
                                     bn1_m, bn1_v, x3b);
  // qkv = x3 @ qkv_w^T   (M=32768,N=768,K=256)
  k_gemm_bf16<<<dim3(6, 256), 256, 0, stream>>>(x3b, qkvw_b, nullptr,
                                                nullptr, qkvb,
                                                32768, 768, 256, 0);
  // fused attention + LePE
  k_attn<<<1024, 256, 0, stream>>>(qkvb, x6f);
  k_lepe<<<32768, 256, 0, stream>>>(qkvb, lepe_w0, lepe_b0, lepe_w1, lepe_b1, x6f);
  k_f2bf<<<(32768*256)/256, 256, 0, stream>>>(x6f, x6b, 32768*256);

  // proj
  k_gemm_bf16<<<dim3(2, 256), 256, 0, stream>>>(x6b, projw_b, proj_b,
                                                projf, nullptr,
                                                32768, 256, 256, 0);
  k_ln256<<<32768, 256, 0, stream>>>(projf, ln1_g, ln1_b, lnb);

  // MLP
  k_gemm_bf16<<<dim3(8, 256), 256, 0, stream>>>(lnb, fc1w_b, fc1_b,
                                                nullptr, fc1o,
                                                32768, 1024, 256, 1);
  k_gemm_bf16<<<dim3(2, 256), 256, 0, stream>>>(fc1o, fc2w_b, fc2_b,
                                                x6ff, nullptr,
                                                32768, 256, 1024, 0);
  // patch merge + LN + reduction
  k_merge_ln<<<8192, 256, 0, stream>>>(x6ff, ln_g, ln_b, x9b);
  k_gemm_bf16<<<dim3(4, 64), 256, 0, stream>>>(x9b, redw_b, red_b,
                                               x11, nullptr,
                                               8192, 512, 1024, 0);
  // deconv + GELU + BN -> output
  k_deconv<<<512, 256, 0, stream>>>(x11, dec_w, dec_b, bn2_g, bn2_b,
                                    bn2_m, bn2_v, (float*)d_out);
}
